// ContrastLoss1_26731876450776
// MI455X (gfx1250) — matrix operand order repaired and checked by reading, not compiled, not run on hardware
//
#include <hip/hip_runtime.h>
#include <hip/hip_bf16.h>

#define TPB 256
#define DD  768
#define POS_SPLIT 4   // blocks per class for the positive pairwise kernel

typedef __attribute__((ext_vector_type(16))) __bf16 v16bf;
typedef __attribute__((ext_vector_type(8)))  float  v8f;

union ABFrag { v16bf v; uint4 q[2]; };

__device__ __forceinline__ unsigned short f2bf(float f) {
  unsigned int u = __float_as_uint(f);
  u += 0x7FFFu + ((u >> 16) & 1u);   // round-to-nearest-even
  return (unsigned short)(u >> 16);
}
__device__ __forceinline__ float bf2f(unsigned short b) {
  return __uint_as_float(((unsigned int)b) << 16);
}

__device__ __forceinline__ float blockReduceSum(float v, float* sh) {
  int tid = threadIdx.x;
  sh[tid] = v;
  __syncthreads();
  for (int s = TPB / 2; s > 0; s >>= 1) {
    if (tid < s) sh[tid] += sh[tid + s];
    __syncthreads();
  }
  float r = sh[0];
  __syncthreads();
  return r;
}

__device__ __forceinline__ void loadFrag(ABFrag& a, ABFrag& b,
                                         const unsigned short* __restrict__ arow,
                                         const unsigned short* __restrict__ brow,
                                         int kb, int h) {
  // A 16x32 bf16: lanes 0-15 hold K = kb+0..7 and kb+16..23; lanes 16-31 shifted by 8
  a.q[0] = *(const uint4*)(arow + kb + h * 8);
  a.q[1] = *(const uint4*)(arow + kb + 16 + h * 8);
  // B 32x16 bf16: same K layout as A (lanes 0-15 K = kb+0..7, kb+16..23; lanes 16-31 shifted by 8)
  b.q[0] = *(const uint4*)(brow + kb + h * 8);
  b.q[1] = *(const uint4*)(brow + kb + 16 + h * 8);
}

// Sum of exp(dot(row_i,row_j)/D) over all ordered pairs (i,j) in [0,n)^2, using
// symmetry: only tile pairs ti<=tj are computed; off-diagonal tiles weighted 2x.
// Tile pairs are strided across (w, W) = (this wave's global index, total waves).
// K loop is double-buffered so next-slab loads overlap the current WMMA.
__device__ float pairwiseExpSumTri(const unsigned short* __restrict__ base, int n,
                                   int w, int W) {
  const int lane = threadIdx.x & 31;
  const int h  = lane >> 4;   // half-wave select
  const int ln = lane & 15;   // A-row / B-col index within tile
  const int T  = (n + 15) >> 4;
  const float invD = 1.0f / (float)DD;
  float s = 0.0f;
  int idx = 0;
  for (int ti = 0; ti < T; ++ti) {
    for (int tj = ti; tj < T; ++tj, ++idx) {
      if (idx % W != w) continue;
      const unsigned short* arow = base + (size_t)(ti * 16 + ln) * DD;
      const unsigned short* brow = base + (size_t)(tj * 16 + ln) * DD;
      ABFrag fa[2], fb[2];
      loadFrag(fa[0], fb[0], arow, brow, 0, h);
      v8f acc = {};
      int buf = 0;
#pragma unroll
      for (int kb = 0; kb < DD; kb += 32) {
        if (kb + 32 < DD) loadFrag(fa[buf ^ 1], fb[buf ^ 1], arow, brow, kb + 32, h);
        acc = __builtin_amdgcn_wmma_f32_16x16x32_bf16(false, fa[buf].v, false, fb[buf].v,
                                                      (short)0, acc, false, false);
        buf ^= 1;
      }
      const float wgt = (ti == tj) ? 1.0f : 2.0f;
#pragma unroll
      for (int r = 0; r < 8; ++r) {
        int m = ti * 16 + r + 8 * h;   // output row
        int c = tj * 16 + ln;          // output col
        if (m < n && c < n) s += wgt * __expf(acc[r] * invD);
      }
    }
  }
  return s;
}

__global__ void k_init(int* counts, int* cursors, float* totPos, float* totNeg, int C) {
  int t = threadIdx.x;
  for (int i = t; i < C; i += blockDim.x) { counts[i] = 0; cursors[i] = 0; }
  if (t == 0) { *totPos = 0.0f; *totNeg = 0.0f; }
}

__global__ void k_count(const int* __restrict__ labels, int* counts, int M) {
  int t = blockIdx.x * blockDim.x + threadIdx.x;
  if (t < M) atomicAdd(&counts[labels[t]], 1);
}

__global__ void k_scan(const int* __restrict__ counts, int* offsets, int C) {
  if (threadIdx.x == 0) {
    int acc = 0;
    for (int c = 0; c < C; ++c) { offsets[c] = acc; acc += counts[c] + 1; }
  }
}

// blocks [0,C): raw char rows -> bucket seed slot. blocks [C,C+M): LN token -> bucket slot.
__global__ void k_ln_scatter(const float* __restrict__ input_f,
                             const float* __restrict__ char_dic,
                             const float* __restrict__ gamma,
                             const float* __restrict__ beta,
                             const int*   __restrict__ labels,
                             const int*   __restrict__ offsets,
                             int* cursors, unsigned short* gathered, int C) {
  __shared__ float sh[TPB];
  __shared__ int slotSh;
  const int b = blockIdx.x, tid = threadIdx.x;
  if (b < C) {
    const float* src = char_dic + (size_t)b * DD;
    unsigned short* dst = gathered + (size_t)offsets[b] * DD;
    for (int d = tid; d < DD; d += TPB) dst[d] = f2bf(src[d]);
    return;
  }
  const int t   = b - C;
  const int lab = labels[t];
  if (tid == 0) {
    int rank = atomicAdd(&cursors[lab], 1);
    slotSh = offsets[lab] + 1 + rank;
  }
  const float* src = input_f + (size_t)t * DD;
  float x0 = src[tid], x1 = src[tid + 256], x2 = src[tid + 512];
  float mu   = blockReduceSum(x0 + x1 + x2, sh) * (1.0f / DD);
  float var  = blockReduceSum(x0*x0 + x1*x1 + x2*x2, sh) * (1.0f / DD) - mu * mu;
  float rstd = rsqrtf(var + 1e-5f);
  unsigned short* dst = gathered + (size_t)slotSh * DD;
#pragma unroll
  for (int i = 0; i < 3; ++i) {
    int d = tid + i * 256;
    dst[d] = f2bf((((i == 0) ? x0 : (i == 1) ? x1 : x2) - mu) * rstd * gamma[d] + beta[d]);
  }
}

// One workgroup per class: bucket mean = (char_dic + token sums) / (count+1).
__global__ void k_mean(const unsigned short* __restrict__ gathered,
                       const int* __restrict__ counts,
                       const int* __restrict__ offsets,
                       float* meanBuf) {
  const int g = blockIdx.x, tid = threadIdx.x;
  const int n = counts[g] + 1;
  const unsigned short* base = gathered + (size_t)offsets[g] * DD;
  for (int d = tid; d < DD; d += TPB) {
    float acc = 0.0f;
    for (int r = 0; r < n; ++r) acc += bf2f(base[(size_t)r * DD + d]);
    meanBuf[(size_t)g * DD + d] = acc / (float)n;
  }
}

// grid (C, POS_SPLIT): masked WMMA pairwise exp-sum per class (positive loss).
__global__ void k_pos(const unsigned short* __restrict__ gathered,
                      const int* __restrict__ counts,
                      const int* __restrict__ offsets,
                      float* totPos) {
  __shared__ float sh[TPB];
  const int g = blockIdx.x;
  const int n = counts[g] + 1;
  const unsigned short* base = gathered + (size_t)offsets[g] * DD;
  const int nwave = TPB >> 5;
  const int w = blockIdx.y * nwave + (threadIdx.x >> 5);
  const int W = gridDim.y * nwave;
  float part = pairwiseExpSumTri(base, n, w, W);
  float bs = blockReduceSum(part, sh);
  if (threadIdx.x == 0) atomicAdd(totPos, bs);
}

// Dictionary update (classes >=1) + LayerNorm -> bf16 rows for the negative path.
__global__ void k_cd(const float* __restrict__ char_dic,
                     const float* __restrict__ gamma,
                     const float* __restrict__ beta,
                     const float* __restrict__ meanBuf,
                     unsigned short* cdbf) {
  __shared__ float sh[TPB];
  const int g = blockIdx.x, tid = threadIdx.x;
  float x[3];
#pragma unroll
  for (int i = 0; i < 3; ++i) {
    int d = tid + i * 256;
    float c = char_dic[(size_t)g * DD + d];
    x[i] = (g >= 1) ? c + 0.1f * meanBuf[(size_t)g * DD + d] : c;
  }
  float mu   = blockReduceSum(x[0] + x[1] + x[2], sh) * (1.0f / DD);
  float var  = blockReduceSum(x[0]*x[0] + x[1]*x[1] + x[2]*x[2], sh) * (1.0f / DD) - mu * mu;
  float rstd = rsqrtf(var + 1e-5f);
#pragma unroll
  for (int i = 0; i < 3; ++i) {
    int d = tid + i * 256;
    cdbf[(size_t)g * DD + d] = f2bf((x[i] - mu) * rstd * gamma[d] + beta[d]);
  }
}

__global__ void k_neg(const unsigned short* __restrict__ cdbf, float* totNeg, int C) {
  __shared__ float sh[TPB];
  float part = pairwiseExpSumTri(cdbf + DD, C - 1, threadIdx.x >> 5, TPB >> 5);
  float bs = blockReduceSum(part, sh);
  if (threadIdx.x == 0) atomicAdd(totNeg, bs);
}

__global__ void k_final(const float* totPos, const float* totNeg, float* out) {
  if (threadIdx.x == 0 && blockIdx.x == 0)
    out[0] = logf(*totNeg) - logf(*totPos);        // -log(pos/neg)
}

extern "C" void kernel_launch(void* const* d_in, const int* in_sizes, int n_in,
                              void* d_out, int out_size, void* d_ws, size_t ws_size,
                              hipStream_t stream) {
  const float* input_f  = (const float*)d_in[0];
  const float* char_dic = (const float*)d_in[1];
  const float* gamma1   = (const float*)d_in[2];
  const float* beta1    = (const float*)d_in[3];
  const int*   target   = (const int*)d_in[4];

  const int M = in_sizes[0] / DD;   // 8192 tokens
  const int C = in_sizes[1] / DD;   // 96 classes

  char* p = (char*)d_ws;
  unsigned short* gathered = (unsigned short*)p; p += (size_t)(C + M + 16) * DD * 2;
  float*          meanBuf  = (float*)p;          p += (size_t)C * DD * 4;
  unsigned short* cdbf     = (unsigned short*)p; p += (size_t)(C + 16) * DD * 2;
  int*            counts   = (int*)p;            p += (size_t)C * 4;
  int*            offsets  = (int*)p;            p += (size_t)C * 4;
  int*            cursors  = (int*)p;            p += (size_t)C * 4;
  float*          totPos   = (float*)p;          p += 4;
  float*          totNeg   = (float*)p;

  k_init<<<1, 128, 0, stream>>>(counts, cursors, totPos, totNeg, C);
  k_count<<<(M + TPB - 1) / TPB, TPB, 0, stream>>>(target, counts, M);
  k_scan<<<1, 32, 0, stream>>>(counts, offsets, C);
  k_ln_scatter<<<C + M, TPB, 0, stream>>>(input_f, char_dic, gamma1, beta1,
                                          target, offsets, cursors, gathered, C);
  k_mean<<<C, TPB, 0, stream>>>(gathered, counts, offsets, meanBuf);
  k_pos<<<dim3(C, POS_SPLIT), TPB, 0, stream>>>(gathered, counts, offsets, totPos);
  k_cd<<<C, TPB, 0, stream>>>(char_dic, gamma1, beta1, meanBuf, cdbf);
  k_neg<<<1, TPB, 0, stream>>>(cdbf, totNeg, C);
  k_final<<<1, 32, 0, stream>>>(totPos, totNeg, (float*)d_out);
}
